// QuantLinearMLX_3307124818123
// MI455X (gfx1250) — matrix operand order repaired and checked by reading, compiled, not run on hardware
//
#include <hip/hip_runtime.h>

// ---- WMMA vector types (wave32, gfx1250) ----
typedef __attribute__((ext_vector_type(16))) _Float16 v16h;
typedef __attribute__((ext_vector_type(8)))  float    v8f;
typedef __attribute__((ext_vector_type(4)))  _Float16 h4;
typedef __attribute__((ext_vector_type(4)))  int      v4i;

#define K_DIM   4096
#define N_DIM   11008
#define PACKED  512      // K_DIM*4/32
#define NGROUPS 32       // K_DIM/128
#define BM      128
#define BN      128
#define BK      64
#define AS_STRIDE 80     // elems; 160B = 5*32B  -> 32B-aligned rows
#define BS_STRIDE 144    // elems; 288B = 9*32B  -> 32B-aligned rows

// ---- gfx1250 async global->LDS path (probe-confirmed to exist; typed per diagnostic) ----
#if defined(__has_builtin)
#  if __has_builtin(__builtin_amdgcn_global_load_async_to_lds_b128)
#    define HAVE_ASYNC_LDS 1
#  endif
#endif
#ifndef HAVE_ASYNC_LDS
#  define HAVE_ASYNC_LDS 0
#endif

#if HAVE_ASYNC_LDS
#  if defined(__has_builtin)
#    if __has_builtin(__builtin_amdgcn_s_wait_asynccnt)
#      define WAIT_ASYNCCNT(n) __builtin_amdgcn_s_wait_asynccnt(n)
#    endif
#  endif
#  ifndef WAIT_ASYNCCNT
#    define WAIT_ASYNCCNT(n) asm volatile("s_wait_asynccnt %0" :: "n"(n) : "memory")
#  endif
#endif

__global__ __launch_bounds__(256)
void q4_gemm_wmma_f16(const float* __restrict__ x,
                      const int*   __restrict__ wpk,
                      const float* __restrict__ scales,
                      const float* __restrict__ biases,
                      const float* __restrict__ bias,
                      float*       __restrict__ out)
{
    __shared__ _Float16 As[BM * AS_STRIDE];   // 20480 B, As[m][k] (f16)
    __shared__ _Float16 Bs[BK * BS_STRIDE];   // 18432 B, Bs[k][n] (f16, dequantized)
    __shared__ int      Wraw[2][BN * 8];      // 2 x 4096 B raw packed int4 tiles

    const int tid  = threadIdx.x;
    const int lane = tid & 31;
    const int wave = tid >> 5;
    const int wm   = wave & 3;    // wave's 32-row M strip
    const int wn   = wave >> 2;   // wave's 64-col N strip

    const int m0 = blockIdx.y * BM;
    const int n0 = blockIdx.x * BN;

    // A staging: thread -> (row = tid>>1, half = tid&1), 32 floats each
    const int ar = tid >> 1;
    const int ah = tid & 1;
    const float* asrc_base = x + (size_t)(m0 + ar) * K_DIM + ah * 32;

    // B staging: thread -> (n_local = tid&127, k-slot = tid>>7); one b128 (4 words = 32 k)
    const int nl   = tid & 127;
    const int kslt = tid >> 7;                       // 0..1
    const int*   bsrc_base = wpk + (size_t)(n0 + nl) * PACKED + kslt * 4;
    const float* s_row = scales + (size_t)(n0 + nl) * NGROUPS;
    const float* z_row = biases + (size_t)(n0 + nl) * NGROUPS;
    int* wraw_slot0 = &Wraw[0][nl * 8 + kslt * 4];
    int* wraw_slot1 = &Wraw[1][nl * 8 + kslt * 4];

    // issue (or perform) the raw-weight tile copy for K-tile k0 into buffer `buf`
    auto copy_w = [&](int k0, int buf) {
        const int* g = bsrc_base + (k0 >> 3);
        int*       l = buf ? wraw_slot1 : wraw_slot0;
#if HAVE_ASYNC_LDS
        __builtin_amdgcn_global_load_async_to_lds_b128(
            (__attribute__((address_space(1))) v4i*)g,
            (__attribute__((address_space(3))) v4i*)l,
            0, 0);
#else
        *(v4i*)l = *(const v4i*)g;
#endif
    };

    v8f acc[2][4];
    #pragma unroll
    for (int i = 0; i < 2; ++i)
        #pragma unroll
        for (int j = 0; j < 4; ++j)
            acc[i][j] = (v8f)(0.0f);

    const int arow = lane & 15;          // A fragment row within 16
    const int ksel = (lane >> 4) * 16;   // A fragment K half select

    // prologue: start async DMA of the first weight tile
    copy_w(0, 0);

    for (int k0 = 0; k0 < K_DIM; k0 += BK) {
        const int buf  = (k0 >> 6) & 1;
        const int more = (k0 + BK < K_DIM);

        // ---------------- stage A (fp32 -> f16 via VGPRs) ----------------
        {
            const float* src = asrc_base + k0;
            _Float16* dst = &As[ar * AS_STRIDE + ah * 32];
            #pragma unroll
            for (int q = 0; q < 8; ++q) {
                float4 v = ((const float4*)src)[q];     // 16B aligned
                h4 t;
                t[0] = (_Float16)v.x; t[1] = (_Float16)v.y;
                t[2] = (_Float16)v.z; t[3] = (_Float16)v.w;
                ((h4*)dst)[q] = t;                       // 8B aligned store
            }
        }

        // kick off the NEXT weight tile DMA before consuming the current one
        if (more) {
            copy_w(k0 + BK, buf ^ 1);
            __builtin_prefetch(asrc_base + (k0 + BK), 0, 1);   // global_prefetch_b8
        }
#if HAVE_ASYNC_LDS
        // per-wave async ops complete in order: after issuing 1 new op,
        // asynccnt<=1 guarantees the current tile's op has landed in LDS.
        if (more) { WAIT_ASYNCCNT(1); } else { WAIT_ASYNCCNT(0); }
#endif

        // ---------------- stage B (LDS raw int4 -> dequant -> f16 LDS) ----------------
        {
            const int g   = k0 >> 7;                     // quant group (BK=64 inside one group)
            const float s  = s_row[g];
            const float zb = z_row[g];
            const int* l = buf ? wraw_slot1 : wraw_slot0;
            v4i w4 = *(const v4i*)l;                     // ds_load_b128 (own async data)
            unsigned int wv[4] = { (unsigned)w4.x, (unsigned)w4.y,
                                   (unsigned)w4.z, (unsigned)w4.w };
            const int kbase = kslt * 32;
            #pragma unroll
            for (int w = 0; w < 4; ++w) {
                unsigned int v = wv[w];
                #pragma unroll
                for (int j = 0; j < 8; ++j) {
                    float q = (float)((v >> (4 * j)) & 0xFu);   // unsigned nibble, per reference
                    Bs[(kbase + w * 8 + j) * BS_STRIDE + nl] = (_Float16)(s * q + zb);
                }
            }
        }

        __syncthreads();

        // ---------------- WMMA compute ----------------
        #pragma unroll
        for (int kk = 0; kk < BK; kk += 32) {
            v16h afrag[2];
            v16h bfrag[4];
            #pragma unroll
            for (int i = 0; i < 2; ++i)
                afrag[i] = *(const v16h*)&As[(wm * 32 + i * 16 + arow) * AS_STRIDE + kk + ksel];
            #pragma unroll
            for (int j = 0; j < 4; ++j)
                for (int s = 0; s < 16; ++s)
                    bfrag[j][s] = Bs[(kk + ksel + s) * BS_STRIDE + wn * 64 + j * 16 + arow];
            #pragma unroll
            for (int i = 0; i < 2; ++i)
                #pragma unroll
                for (int j = 0; j < 4; ++j)
                    acc[i][j] = __builtin_amdgcn_wmma_f32_16x16x32_f16(
                        /*neg_a=*/false, afrag[i],
                        /*neg_b=*/false, bfrag[j],
                        /*c_mod=*/(short)0, acc[i][j],
                        /*reuse_a=*/false, /*reuse_b=*/false);
        }

        __syncthreads();
    }

    // ---------------- epilogue: add bias, store fp32 ----------------
    // C/D layout: lanes 0-15 -> N=lane, M=r ; lanes 16-31 -> N=lane-16, M=r+8
    #pragma unroll
    for (int i = 0; i < 2; ++i) {
        const int mrow = m0 + wm * 32 + i * 16 + (lane >> 4) * 8;
        #pragma unroll
        for (int j = 0; j < 4; ++j) {
            const int col = n0 + wn * 64 + j * 16 + (lane & 15);
            const float bv = bias[col];
            #pragma unroll
            for (int r = 0; r < 8; ++r) {
                out[(size_t)(mrow + r) * N_DIM + col] = acc[i][j][r] + bv;
            }
        }
    }
}

extern "C" void kernel_launch(void* const* d_in, const int* in_sizes, int n_in,
                              void* d_out, int out_size, void* d_ws, size_t ws_size,
                              hipStream_t stream) {
    const float* x      = (const float*)d_in[0];   // (4,2048,4096) f32
    const int*   wpk    = (const int*)  d_in[1];   // (11008,512)  i32 packed int4
    const float* scales = (const float*)d_in[2];   // (11008,32)
    const float* biases = (const float*)d_in[3];   // (11008,32)
    const float* bias   = (const float*)d_in[4];   // (11008,)

    float* out = (float*)d_out;

    const int M = in_sizes[0] / K_DIM;             // 8192
    dim3 grid(N_DIM / BN, M / BM);                 // (86, 64)
    q4_gemm_wmma_f16<<<grid, 256, 0, stream>>>(x, wpk, scales, biases, bias, out);
}
